// DiffFormerBlock_67053029425606
// MI455X (gfx1250) — matrix operand order repaired and checked by reading, not compiled, not run on hardware
//
#include <hip/hip_runtime.h>
#include <hip/hip_bf16.h>

typedef __attribute__((ext_vector_type(16))) _Float16 v16h;
typedef __attribute__((ext_vector_type(8)))  _Float16 v8h;
typedef __attribute__((ext_vector_type(8)))  float    v8f;

namespace {

constexpr int   kC  = 384;
constexpr int   kNH = 12;
constexpr int   kHD = 32;
constexpr int   kNT = 64;      // tokens per window
constexpr int   kNW = 512;     // windows
constexpr int   kM  = 32768;   // total rows (= 512*64 = 8*4096)
constexpr float kScale = 0.17677669529663687f;  // 32^-0.5

union Frag { v16h v; v8h h[2]; };

// A-matrix 16x32 f16 fragment (CDNA5 ISA 7.12.2): lane r holds row M=r;
// halves 0..7 = K { hi*8 .. +7 }, halves 8..15 = K { 16+hi*8 .. +7 }
__device__ inline v16h frag_row_a(const _Float16* rowp, int k, int hi) {
  Frag u;
  const _Float16* p = rowp + k + hi * 8;
  u.h[0] = *(const v8h*)p;
  u.h[1] = *(const v8h*)(p + 16);
  return u.v;
}

// B-matrix 32x16 f16 fragment: lane r = column N=r; halves 0..7 = K { hi*8 .. +7 },
// halves 8..15 = K { 16+hi*8 .. +7 }. rowp points at B^T row (a column of B).
__device__ inline v16h frag_row_b(const _Float16* rowp, int k, int hi) {
  Frag u;
  const _Float16* p = rowp + k + hi * 8;
  u.h[0] = *(const v8h*)p;
  u.h[1] = *(const v8h*)(p + 16);
  return u.v;
}

__device__ inline v16h frag_a(const _Float16* base, int stride, int row, int k, int hi) {
  return frag_row_a(base + (size_t)row * stride, k, hi);
}
__device__ inline v16h frag_b(const _Float16* base, int stride, int col, int k, int hi) {
  return frag_row_b(base + (size_t)col * stride, k, hi);
}

#define MODE_F16    0
#define MODE_GELU   1
#define MODE_Q      2
#define MODE_KV     3
#define MODE_REVRES 4
#define MODE_ACC    5

__device__ inline void store_out(int mode, int row, int col, float v, int N,
                                 _Float16* __restrict__ outH, _Float16* __restrict__ outH2,
                                 float* __restrict__ outF, const float* __restrict__ qg,
                                 const float* __restrict__ resid)
{
  if (mode == MODE_F16) {
    outH[(size_t)row * N + col] = (_Float16)v;
  } else if (mode == MODE_GELU) {
    v = 0.5f * v * (1.0f + erff(v * 0.70710678118654752f));
    outH[(size_t)row * N + col] = (_Float16)v;
  } else if (mode == MODE_Q) {
    // out: [win][head][tok][d]; global query folded in; pre-scaled by HD^-1/2
    const int win = row >> 6, n = row & 63;
    const int h = col >> 5, d = col & 31;
    if (qg) v += qg[(((size_t)(win >> 6) * kNH + h) * kNT + n) * kHD + d];
    v *= kScale;
    outH[(((size_t)(win * kNH + h)) * kNT + n) * kHD + d] = (_Float16)v;
  } else if (mode == MODE_KV) {
    const int win = row >> 6, n = row & 63;
    if (col < kC) {                 // K: [win][head][tok][d]
      const int h = col >> 5, d = col & 31;
      outH[(((size_t)(win * kNH + h)) * kNT + n) * kHD + d] = (_Float16)v;
    } else {                        // V transposed: [win][head][d][tok]
      const int c2 = col - kC;
      const int h = c2 >> 5, d = c2 & 31;
      outH2[(((size_t)(win * kNH + h)) * kHD + d) * kNT + n] = (_Float16)v;
    }
  } else if (mode == MODE_REVRES) {
    // window_reverse + residual into f32 output
    const int win = row >> 6, n = row & 63;
    const int b = win >> 6, wr = (win >> 3) & 7, wc = win & 7;
    const int hs = n >> 3, wsx = n & 7;
    const size_t idx =
        (((size_t)b * 4096) + (size_t)(wr * 8 + hs) * 64 + (wc * 8 + wsx)) * kC + col;
    outF[idx] = v + resid[idx];
  } else { // MODE_ACC
    const size_t idx = (size_t)row * N + col;
    outF[idx] += v;
  }
}

} // namespace

#define WMMA_F16(a, b, c) \
  __builtin_amdgcn_wmma_f32_16x16x32_f16(false, (a), false, (b), (short)0, (c), false, false)

// ---------------------------------------------------------------------------
// Generic f16 WMMA GEMM: out = A[M,K] @ B[K,N] + bias, fused epilogues.
// Block = 4 waves; each wave computes a 32(M) x 64(N) strip (8 WMMA / k-step).
// K loop is software-pipelined: fragments for k+32 are loaded into fresh
// registers while the 8 WMMAs for step k execute.
// ---------------------------------------------------------------------------
__global__ __launch_bounds__(128)
void gemm_wmma(const _Float16* __restrict__ A, const _Float16* __restrict__ Bt,
               const float* __restrict__ bias, int M, int N, int K, int mode,
               _Float16* __restrict__ outH, _Float16* __restrict__ outH2,
               float* __restrict__ outF, const float* __restrict__ qg,
               const float* __restrict__ resid)
{
  const int wv   = threadIdx.x >> 5;
  const int lane = threadIdx.x & 31;
  const int r    = lane & 15;
  const int hi   = (lane >> 4) & 1;
  const int rowBase = blockIdx.x * 128 + wv * 32;
  const int colBase = blockIdx.y * 64;

  const _Float16* a0p = A + (size_t)(rowBase + r) * K;
  const _Float16* a1p = A + (size_t)(rowBase + 16 + r) * K;
  const _Float16* b0p = Bt + (size_t)(colBase + r) * K;
  const _Float16* b1p = Bt + (size_t)(colBase + 16 + r) * K;
  const _Float16* b2p = Bt + (size_t)(colBase + 32 + r) * K;
  const _Float16* b3p = Bt + (size_t)(colBase + 48 + r) * K;

  v8f z = {};
  v8f acc[8] = { z, z, z, z, z, z, z, z };

  // prologue: fragments for k = 0
  v16h a0 = frag_row_a(a0p, 0, hi);
  v16h a1 = frag_row_a(a1p, 0, hi);
  v16h b0 = frag_row_b(b0p, 0, hi);
  v16h b1 = frag_row_b(b1p, 0, hi);
  v16h b2 = frag_row_b(b2p, 0, hi);
  v16h b3 = frag_row_b(b3p, 0, hi);

  for (int k = 32; k < K; k += 32) {
    // issue next-step loads first so they overlap this step's WMMAs
    v16h a0n = frag_row_a(a0p, k, hi);
    v16h a1n = frag_row_a(a1p, k, hi);
    v16h b0n = frag_row_b(b0p, k, hi);
    v16h b1n = frag_row_b(b1p, k, hi);
    v16h b2n = frag_row_b(b2p, k, hi);
    v16h b3n = frag_row_b(b3p, k, hi);
    __builtin_prefetch(a0p + k + 224, 0, 1);   // global_prefetch_b8, A stream ahead

    acc[0] = WMMA_F16(a0, b0, acc[0]);
    acc[1] = WMMA_F16(a0, b1, acc[1]);
    acc[2] = WMMA_F16(a0, b2, acc[2]);
    acc[3] = WMMA_F16(a0, b3, acc[3]);
    acc[4] = WMMA_F16(a1, b0, acc[4]);
    acc[5] = WMMA_F16(a1, b1, acc[5]);
    acc[6] = WMMA_F16(a1, b2, acc[6]);
    acc[7] = WMMA_F16(a1, b3, acc[7]);

    a0 = a0n; a1 = a1n; b0 = b0n; b1 = b1n; b2 = b2n; b3 = b3n;
  }
  // epilogue step
  acc[0] = WMMA_F16(a0, b0, acc[0]);
  acc[1] = WMMA_F16(a0, b1, acc[1]);
  acc[2] = WMMA_F16(a0, b2, acc[2]);
  acc[3] = WMMA_F16(a0, b3, acc[3]);
  acc[4] = WMMA_F16(a1, b0, acc[4]);
  acc[5] = WMMA_F16(a1, b1, acc[5]);
  acc[6] = WMMA_F16(a1, b2, acc[6]);
  acc[7] = WMMA_F16(a1, b3, acc[7]);

#pragma unroll
  for (int hf = 0; hf < 2; ++hf) {
#pragma unroll
    for (int t = 0; t < 4; ++t) {
      const int col = colBase + t * 16 + r;
      const float bb = bias[col];
#pragma unroll
      for (int j = 0; j < 8; ++j) {
        const int row = rowBase + hf * 16 + j + hi * 8;  // C/D: VGPR j -> M = j + 8*hi
        store_out(mode, row, col, acc[hf * 4 + t][j] + bb, N,
                  outH, outH2, outF, qg, resid);
      }
    }
  }
}

// ---------------------------------------------------------------------------
// Window attention core: one block per (window, head).
// S = Q_scaled @ K^T + relposbias; softmax; out = P @ V.
// ---------------------------------------------------------------------------
__global__ __launch_bounds__(128)
void attn_wmma(const _Float16* __restrict__ Q, const _Float16* __restrict__ Kb,
               const _Float16* __restrict__ Vt, const float* __restrict__ rpb,
               _Float16* __restrict__ AO)
{
  __shared__ __align__(16) float    sS[64][64];
  __shared__ __align__(16) _Float16 sP[64][64];

  const int win  = blockIdx.x / kNH;
  const int h    = blockIdx.x % kNH;
  const int wv   = threadIdx.x >> 5;
  const int lane = threadIdx.x & 31;
  const int r    = lane & 15;
  const int hi   = (lane >> 4) & 1;
  const int rowBase = wv * 16;

  const _Float16* Qp = Q  + (size_t)(win * kNH + h) * kNT * kHD;  // [64][32]
  const _Float16* Kp = Kb + (size_t)(win * kNH + h) * kNT * kHD;  // [64][32]
  const _Float16* Vp = Vt + (size_t)(win * kNH + h) * kHD * kNT;  // [32][64] (V^T)

  // ---- S = Q @ K^T (K-dim = 32 -> one WMMA per 16x16 tile) ----
  v16h a  = frag_a(Qp, kHD, rowBase + r, 0, hi);
  v16h kb0 = frag_b(Kp, kHD,  0 + r, 0, hi);
  v16h kb1 = frag_b(Kp, kHD, 16 + r, 0, hi);
  v16h kb2 = frag_b(Kp, kHD, 32 + r, 0, hi);
  v16h kb3 = frag_b(Kp, kHD, 48 + r, 0, hi);
  v8f z = {};
  v8f sacc[4];
  sacc[0] = WMMA_F16(a, kb0, z);
  sacc[1] = WMMA_F16(a, kb1, z);
  sacc[2] = WMMA_F16(a, kb2, z);
  sacc[3] = WMMA_F16(a, kb3, z);

#pragma unroll
  for (int nt = 0; nt < 4; ++nt) {
    const int m = nt * 16 + r;
#pragma unroll
    for (int j = 0; j < 8; ++j) {
      const int n = rowBase + j + hi * 8;
      const int idx = ((n >> 3) - (m >> 3) + 7) * 15 + ((n & 7) - (m & 7) + 7);
      sS[n][m] = sacc[nt][j] + rpb[idx * kNH + h];
    }
  }
  __syncthreads();

  // ---- softmax over keys (row-per-thread, 2-pass) ----
  if (threadIdx.x < 64) {
    const int n = threadIdx.x;
    float mx = -1e30f;
    for (int m = 0; m < 64; ++m) mx = fmaxf(mx, sS[n][m]);
    float s = 0.0f;
    for (int m = 0; m < 64; ++m) s += __expf(sS[n][m] - mx);
    const float inv = 1.0f / s;
    for (int m = 0; m < 64; ++m) sP[n][m] = (_Float16)(__expf(sS[n][m] - mx) * inv);
  }
  __syncthreads();

  // ---- out = P[64,64] @ V[64,32] (K-dim 64 -> 2 WMMA per tile) ----
  v16h pa0 = frag_a(&sP[0][0], 64, rowBase + r, 0, hi);
  v16h pa1 = frag_a(&sP[0][0], 64, rowBase + r, 32, hi);
#pragma unroll
  for (int dt = 0; dt < 2; ++dt) {
    v16h vb0 = frag_b(Vp, kNT, dt * 16 + r, 0, hi);
    v16h vb1 = frag_b(Vp, kNT, dt * 16 + r, 32, hi);
    v8f c = {};
    c = WMMA_F16(pa0, vb0, c);
    c = WMMA_F16(pa1, vb1, c);
#pragma unroll
    for (int j = 0; j < 8; ++j) {
      const int row = rowBase + j + hi * 8;
      const int d = dt * 16 + r;
      AO[((size_t)(win * kNT) + row) * kC + h * kHD + d] = (_Float16)c[j];
    }
  }
}

// ---------------------------------------------------------------------------
// LayerNorm over C=384; optional window-partition permutation of output rows.
// One block (128 threads) per token.
// ---------------------------------------------------------------------------
__global__ __launch_bounds__(128)
void ln_rows(const float* __restrict__ x, const float* __restrict__ g,
             const float* __restrict__ bta, _Float16* __restrict__ out, int windowed)
{
  __shared__ float red[128], red2[128];
  const int tk = blockIdx.x;
  const int t  = threadIdx.x;
  const float* row = x + (size_t)tk * kC;

  float v[3], s = 0.0f, s2 = 0.0f;
#pragma unroll
  for (int i = 0; i < 3; ++i) { v[i] = row[t + i * 128]; s += v[i]; s2 += v[i] * v[i]; }
  red[t] = s; red2[t] = s2; __syncthreads();
  for (int o = 64; o > 0; o >>= 1) {
    if (t < o) { red[t] += red[t + o]; red2[t] += red2[t + o]; }
    __syncthreads();
  }
  const float mean = red[0] * (1.0f / kC);
  const float var  = red2[0] * (1.0f / kC) - mean * mean;
  const float inv  = rsqrtf(var + 1e-5f);

  size_t orow;
  if (windowed) {
    const int b = tk >> 12, p = tk & 4095, hh = p >> 6, ww = p & 63;
    const int win = ((b << 3) + (hh >> 3)) * 8 + (ww >> 3);
    const int n = ((hh & 7) << 3) + (ww & 7);
    orow = (size_t)win * kNT + n;
  } else {
    orow = (size_t)tk;
  }
  _Float16* op = out + orow * kC;
#pragma unroll
  for (int i = 0; i < 3; ++i) {
    const int c = t + i * 128;
    op[c] = (_Float16)((v[i] - mean) * inv * g[c] + bta[c]);
  }
}

__global__ void absdiff_h(const _Float16* __restrict__ a, const _Float16* __restrict__ b,
                          _Float16* __restrict__ o, int n)
{
  const int i = blockIdx.x * blockDim.x + threadIdx.x;
  if (i < n) {
    const float d = (float)a[i] - (float)b[i];
    o[i] = (_Float16)fabsf(d);
  }
}

__global__ void transpose_w(const float* __restrict__ src, _Float16* __restrict__ dst,
                            int K, int N)
{
  const int i = blockIdx.x * blockDim.x + threadIdx.x;
  if (i < K * N) {
    const int k = i / N, n = i % N;
    dst[(size_t)n * K + k] = (_Float16)src[i];
  }
}

// ---------------------------------------------------------------------------
extern "C" void kernel_launch(void* const* d_in, const int* in_sizes, int n_in,
                              void* d_out, int out_size, void* d_ws, size_t ws_size,
                              hipStream_t stream)
{
  (void)in_sizes; (void)n_in; (void)out_size; (void)ws_size;
  auto fin = [&](int i) { return (const float*)d_in[i]; };

  const float* x  = fin(0);
  const float* y  = fin(1);
  const float* xq = fin(2);          // [B,1,NH,64,32]
  const float* yq = fin(3);
  // LN params: n11 (4,5) n12 (6,7) n21 (8,9) n22 (10,11)
  // attn blocks a1,a2,g1,g2 at base 12 + 7*i: qw,qb,kvw,kvb,pw,pb,rpb
  // mlp: m1_w1 40, m1_b1 41, m1_w2 42, m1_b2 43, m2_* 44..47

  // ---- workspace carve-up (sequential 256B-aligned) ----
  char* ws = (char*)d_ws;
  size_t off = 0;
  auto take = [&](size_t elems) -> _Float16* {
    _Float16* p = (_Float16*)(ws + off);
    off += ((elems * sizeof(_Float16) + 255) / 256) * 256;
    return p;
  };
  _Float16 *qwT[4], *kvwT[4], *pwT[4];
  for (int i = 0; i < 4; ++i) {
    qwT[i]  = take((size_t)kC * kC);
    kvwT[i] = take((size_t)kC * 2 * kC);
    pwT[i]  = take((size_t)kC * kC);
  }
  _Float16* w1T[2]; _Float16* w2T[2];
  w1T[0] = take((size_t)kC * 4 * kC); w2T[0] = take((size_t)kC * 4 * kC);
  w1T[1] = take((size_t)kC * 4 * kC); w2T[1] = take((size_t)kC * 4 * kC);

  const size_t ACT = (size_t)kM * kC;   // 12.58M f16 elems per activation buffer
  _Float16* xw = take(ACT);
  _Float16* yw = take(ACT);
  _Float16* fw = take(ACT);
  _Float16* Qb = take(ACT);   // Qb,Kb,Vb,AO contiguous -> aliased by MLP hidden
  _Float16* Kb = take(ACT);
  _Float16* Vb = take(ACT);
  _Float16* AO = take(ACT);
  _Float16* X1 = take(ACT);
  _Float16* Y1 = take(ACT);
  _Float16* F1  = xw;                          // reuse after stage 1
  _Float16* LNb = yw;                          // reuse after stage 1
  _Float16* Hm  = Qb;                          // 4*ACT == kM*1536, reuse after stage 2

  float* outX = (float*)d_out;
  float* outY = outX + ACT;

  // ---- 1. transpose/convert weights to f16 [N,K] ----
  auto tw = [&](const float* s, _Float16* d, int K, int N) {
    transpose_w<<<(K * N + 255) / 256, 256, 0, stream>>>(s, d, K, N);
  };
  for (int i = 0; i < 4; ++i) {
    const int base = 12 + i * 7;
    tw(fin(base + 0), qwT[i],  kC, kC);
    tw(fin(base + 2), kvwT[i], kC, 2 * kC);
    tw(fin(base + 4), pwT[i],  kC, kC);
  }
  tw(fin(40), w1T[0], kC, 4 * kC);  tw(fin(42), w2T[0], 4 * kC, kC);
  tw(fin(44), w1T[1], kC, 4 * kC);  tw(fin(46), w2T[1], 4 * kC, kC);

  // ---- 2. LN + window partition, diff feature ----
  ln_rows<<<kM, 128, 0, stream>>>(x, fin(4), fin(5), xw, 1);
  ln_rows<<<kM, 128, 0, stream>>>(y, fin(6), fin(7), yw, 1);
  absdiff_h<<<((int)ACT + 255) / 256, 256, 0, stream>>>(xw, yw, fw, (int)ACT);

  const dim3 blk(128);
  auto gemm = [&](const _Float16* A, const _Float16* Bt, const float* bias,
                  int N, int K, int mode, _Float16* oh, _Float16* oh2,
                  float* of, const float* qg, const float* resid) {
    dim3 grid(kM / 128, N / 64);
    gemm_wmma<<<grid, blk, 0, stream>>>(A, Bt, bias, kM, N, K, mode,
                                        oh, oh2, of, qg, resid);
  };
  auto attn = [&](const float* rpb) {
    attn_wmma<<<kNW * kNH, blk, 0, stream>>>(Qb, Kb, Vb, rpb, AO);
  };

  // ---- 3. stage 1: x (a1: q from xw, kv from fw) ----
  gemm(xw, qwT[0],  fin(13), kC,     kC, MODE_Q,  Qb, nullptr, nullptr, nullptr, nullptr);
  gemm(fw, kvwT[0], fin(15), 2 * kC, kC, MODE_KV, Kb, Vb,      nullptr, nullptr, nullptr);
  attn(fin(18));
  gemm(AO, pwT[0],  fin(17), kC,     kC, MODE_F16, X1, nullptr, nullptr, nullptr, nullptr);
  //          stage 1: y (a2: q from yw, kv from fw)
  gemm(yw, qwT[1],  fin(20), kC,     kC, MODE_Q,  Qb, nullptr, nullptr, nullptr, nullptr);
  gemm(fw, kvwT[1], fin(22), 2 * kC, kC, MODE_KV, Kb, Vb,      nullptr, nullptr, nullptr);
  attn(fin(25));
  gemm(AO, pwT[1],  fin(24), kC,     kC, MODE_F16, Y1, nullptr, nullptr, nullptr, nullptr);

  // ---- 4. f1 = |x1 - y1| ----
  absdiff_h<<<((int)ACT + 255) / 256, 256, 0, stream>>>(X1, Y1, F1, (int)ACT);

  // ---- 5. stage 2: x (g1: q from x1 + x_q, kv from f1) -> reverse+residual ----
  gemm(X1, qwT[2],  fin(27), kC,     kC, MODE_Q,  Qb, nullptr, nullptr, xq, nullptr);
  gemm(F1, kvwT[2], fin(29), 2 * kC, kC, MODE_KV, Kb, Vb,      nullptr, nullptr, nullptr);
  attn(fin(32));
  gemm(AO, pwT[2],  fin(31), kC,     kC, MODE_REVRES, nullptr, nullptr, outX, nullptr, x);
  //          stage 2: y (g2)
  gemm(Y1, qwT[3],  fin(34), kC,     kC, MODE_Q,  Qb, nullptr, nullptr, yq, nullptr);
  gemm(F1, kvwT[3], fin(36), 2 * kC, kC, MODE_KV, Kb, Vb,      nullptr, nullptr, nullptr);
  attn(fin(39));
  gemm(AO, pwT[3],  fin(38), kC,     kC, MODE_REVRES, nullptr, nullptr, outY, nullptr, y);

  // ---- 6. MLP x: xo += gelu(LN(xo) @ w1 + b1) @ w2 + b2 ----
  ln_rows<<<kM, 128, 0, stream>>>(outX, fin(8), fin(9), LNb, 0);
  gemm(LNb, w1T[0], fin(41), 4 * kC, kC,     MODE_GELU, Hm, nullptr, nullptr, nullptr, nullptr);
  gemm(Hm,  w2T[0], fin(43), kC,     4 * kC, MODE_ACC,  nullptr, nullptr, outX, nullptr, nullptr);
  //          MLP y
  ln_rows<<<kM, 128, 0, stream>>>(outY, fin(10), fin(11), LNb, 0);
  gemm(LNb, w1T[1], fin(45), 4 * kC, kC,     MODE_GELU, Hm, nullptr, nullptr, nullptr, nullptr);
  gemm(Hm,  w2T[1], fin(47), kC,     4 * kC, MODE_ACC,  nullptr, nullptr, outY, nullptr, nullptr);
}
